// RandLinear_15109694947761
// MI455X (gfx1250) — compile-verified, matrix operand order checked by reading, not run on hardware
//
#include <hip/hip_runtime.h>
#include <cstdint>
#include <cstddef>

#define IN_DIM  4096
#define OUT_DIM 4096
#define NB      8192

typedef __attribute__((ext_vector_type(16))) __bf16 v16bf;
typedef __attribute__((ext_vector_type(8)))  float  v8f;
typedef __attribute__((ext_vector_type(4)))  __bf16 v4bf;
typedef __attribute__((vector_size(16)))     int    vi4;   // 128-bit payload type

struct U32x8 { uint4 lo, hi; };

constexpr int BM = 128, BN = 128, BK = 32;
constexpr int LDSS = BK + 8;            // padded LDS row stride (bf16 elems) -> 80B
constexpr int KT = IN_DIM / BK;         // 128 k-steps

#ifndef __has_builtin
#define __has_builtin(x) 0
#endif
#if __has_builtin(__builtin_amdgcn_global_load_async_to_lds_b128) && \
    __has_builtin(__builtin_amdgcn_s_wait_asynccnt)
#define USE_ASYNC_LDS 1
#else
#define USE_ASYNC_LDS 0
#endif

// 16-byte global -> LDS copy: async data-mover path on CDNA5, VGPR staging fallback
__device__ __forceinline__ void cp16_g2l(__bf16* l, const __bf16* g) {
#if USE_ASYNC_LDS
    __builtin_amdgcn_global_load_async_to_lds_b128(
        (__attribute__((address_space(1))) vi4*)(uintptr_t)g,
        (__attribute__((address_space(3))) vi4*)(uintptr_t)l,
        /*offset=*/0, /*cpol=*/0);
#else
    *(uint4*)l = *(const uint4*)g;
#endif
}

__device__ __forceinline__ void cp_wait() {
#if USE_ASYNC_LDS
    __builtin_amdgcn_s_wait_asynccnt(0);
#endif
}

// ---------------- prep kernels (memory-bound, trivial) ----------------

__global__ void prep_weight_kernel(const float* __restrict__ mu,
                                   const float* __restrict__ ls,
                                   const float* __restrict__ ep,
                                   __bf16* __restrict__ wb) {
    int i = blockIdx.x * blockDim.x + threadIdx.x;      // index over float4s
    float4 m = ((const float4*)mu)[i];
    float4 l = ((const float4*)ls)[i];
    float4 e = ((const float4*)ep)[i];
    v4bf r;
    r[0] = (__bf16)(m.x + __expf(l.x) * e.x);
    r[1] = (__bf16)(m.y + __expf(l.y) * e.y);
    r[2] = (__bf16)(m.z + __expf(l.z) * e.z);
    r[3] = (__bf16)(m.w + __expf(l.w) * e.w);
    ((v4bf*)wb)[i] = r;
}

__global__ void prep_x_kernel(const float* __restrict__ x,
                              __bf16* __restrict__ xb) {
    int i = blockIdx.x * blockDim.x + threadIdx.x;
    float4 v = ((const float4*)x)[i];
    v4bf r;
    r[0] = (__bf16)v.x; r[1] = (__bf16)v.y; r[2] = (__bf16)v.z; r[3] = (__bf16)v.w;
    ((v4bf*)xb)[i] = r;
}

__global__ void prep_bias_kernel(const float* __restrict__ mu,
                                 const float* __restrict__ ls,
                                 const float* __restrict__ ep,
                                 float* __restrict__ b) {
    int i = blockIdx.x * blockDim.x + threadIdx.x;
    b[i] = mu[i] + __expf(ls[i]) * ep[i];
}

// ---------------- bf16 WMMA GEMM:  out[m,n] = sum_k Xb[m,k]*Wb[n,k] + bias[n] ----------------

__device__ __forceinline__ v16bf load_frag(const __bf16* p) {
    // lane covers row (M or N) = lane%16 ; K = fk..fk+7 and fk+16..fk+23, fk = (lane/16)*8
    U32x8 u;
    u.lo = *(const uint4*)p;          // K = fk .. fk+7
    u.hi = *(const uint4*)(p + 16);   // K = fk+16 .. fk+23
    return __builtin_bit_cast(v16bf, u);
}

__global__ __launch_bounds__(256)
void RandLinear_gemm_kernel(const __bf16* __restrict__ Xb,
                            const __bf16* __restrict__ Wb,
                            const float* __restrict__ bias,
                            float* __restrict__ out) {
    __shared__ __bf16 As[2][BM * LDSS];
    __shared__ __bf16 Bs[2][BN * LDSS];

    const int t    = threadIdx.x;
    const int lane = t & 31;
    const int wave = t >> 5;
    const int wm   = (wave >> 2) * 64;      // wave M offset in block tile: 0 / 64
    const int wn   = (wave & 3) * 32;       // wave N offset: 0 / 32 / 64 / 96

    const int bm = blockIdx.y * BM;         // row block  (batch dim, 8192)
    const int bn = blockIdx.x * BN;         // col block  (OUT dim, 4096)

    // cooperative global->LDS staging coords: 2 x b128 per matrix per thread per stage
    const int r0 = t >> 2;                  // 0..63   (rows r0 and r0+64)
    const int c8 = (t & 3) * 8;             // 0,8,16,24 (bf16 elems)

    const __bf16* gA = Xb + (size_t)(bm + r0) * IN_DIM + c8;
    const __bf16* gB = Wb + (size_t)(bn + r0) * IN_DIM + c8;

    // fragment lane coords (identical pattern for A and B in NT layout)
    const int fm = lane & 15;
    const int fk = (lane >> 4) * 8;

    v8f acc[4][2];
    const v8f vzero = {0.f, 0.f, 0.f, 0.f, 0.f, 0.f, 0.f, 0.f};
#pragma unroll
    for (int i = 0; i < 4; ++i)
#pragma unroll
        for (int j = 0; j < 2; ++j) acc[i][j] = vzero;

    // preload stage 0
#pragma unroll
    for (int s = 0; s < 2; ++s) {
        cp16_g2l(&As[0][(r0 + s * 64) * LDSS + c8], gA + (size_t)s * 64 * IN_DIM);
        cp16_g2l(&Bs[0][(r0 + s * 64) * LDSS + c8], gB + (size_t)s * 64 * IN_DIM);
    }
    cp_wait();
    __syncthreads();

    for (int kt = 0; kt < KT; ++kt) {
        const int cur = kt & 1;
        const bool hn = (kt + 1) < KT;

        if (hn) {
            // fire async copies for stage kt+1 into the other buffer; they
            // overlap the WMMAs below (data mover writes LDS, no VGPRs used)
            const int nxt = cur ^ 1;
            const int ko  = (kt + 1) * BK;
#pragma unroll
            for (int s = 0; s < 2; ++s) {
                cp16_g2l(&As[nxt][(r0 + s * 64) * LDSS + c8],
                         gA + (size_t)s * 64 * IN_DIM + ko);
                cp16_g2l(&Bs[nxt][(r0 + s * 64) * LDSS + c8],
                         gB + (size_t)s * 64 * IN_DIM + ko);
            }
            if (kt + 4 < KT) {   // pull ~4 stages ahead into L2
                const int kp = (kt + 4) * BK;
                __builtin_prefetch(gA + kp, 0, 1);
                __builtin_prefetch(gB + kp, 0, 1);
            }
        }

        // fragments from LDS[cur]
        v16bf af[4], bfv[2];
#pragma unroll
        for (int i = 0; i < 4; ++i)
            af[i] = load_frag(&As[cur][(wm + i * 16 + fm) * LDSS + fk]);
#pragma unroll
        for (int j = 0; j < 2; ++j)
            bfv[j] = load_frag(&Bs[cur][(wn + j * 16 + fm) * LDSS + fk]);

#pragma unroll
        for (int i = 0; i < 4; ++i)
#pragma unroll
            for (int j = 0; j < 2; ++j)
                acc[i][j] = __builtin_amdgcn_wmma_f32_16x16x32_bf16(
                    /*neg_a=*/false, af[i], /*neg_b=*/false, bfv[j],
                    /*c_mod=*/(short)0, acc[i][j],
                    /*reuse_a=*/false, /*reuse_b=*/false);

        if (hn) cp_wait();       // our async writes for stage kt+1 are in LDS
        __syncthreads();         // everyone's writes visible, reads of cur done
    }

    // epilogue: C/D layout -> M = vgpr + 8*(lane/16), N = lane%16
    const int mhi = (lane >> 4) * 8;
#pragma unroll
    for (int j = 0; j < 2; ++j) {
        const int gn = bn + wn + j * 16 + fm;
        const float bj = bias[gn];
#pragma unroll
        for (int i = 0; i < 4; ++i) {
            const int gm = bm + wm + i * 16 + mhi;
            float* po = out + (size_t)gm * OUT_DIM + gn;
#pragma unroll
            for (int r = 0; r < 8; ++r)
                po[(size_t)r * OUT_DIM] = acc[i][j][r] + bj;
        }
    }
}

// ---------------- launch ----------------

extern "C" void kernel_launch(void* const* d_in, const int* in_sizes, int n_in,
                              void* d_out, int out_size, void* d_ws, size_t ws_size,
                              hipStream_t stream) {
    const float* x   = (const float*)d_in[0];
    const float* muw = (const float*)d_in[1];
    const float* lsw = (const float*)d_in[2];
    const float* mub = (const float*)d_in[3];
    const float* lsb = (const float*)d_in[4];
    const float* epw = (const float*)d_in[5];
    const float* epb = (const float*)d_in[6];
    float* out = (float*)d_out;

    char* ws = (char*)d_ws;
    __bf16* Wb  = (__bf16*)ws;                                              // 32 MiB
    __bf16* Xbf = (__bf16*)(ws + (size_t)OUT_DIM * IN_DIM * 2);             // 64 MiB
    float*  bia = (float*)(ws + (size_t)OUT_DIM * IN_DIM * 2
                              + (size_t)NB * IN_DIM * 2);                   // 16 KiB

    prep_weight_kernel<<<(OUT_DIM * (size_t)IN_DIM / 4) / 256, 256, 0, stream>>>(muw, lsw, epw, Wb);
    prep_x_kernel<<<((size_t)NB * IN_DIM / 4) / 256, 256, 0, stream>>>(x, Xbf);
    prep_bias_kernel<<<OUT_DIM / 256, 256, 0, stream>>>(mub, lsb, epb, bia);

    dim3 grid(OUT_DIM / BN, NB / BM);
    RandLinear_gemm_kernel<<<grid, 256, 0, stream>>>(Xbf, Wb, bia, out);
}
